// GroupedQueryAttention_64381559767959
// MI455X (gfx1250) — matrix operand order repaired and checked by reading, not compiled, not run on hardware
//
#include <hip/hip_runtime.h>

// ---- problem constants (fixed by the reference module) ----
constexpr int Bb   = 8;
constexpr int Tt   = 16;
constexpr int SPc  = 4080;
constexpr int DIMc = 4096;
constexpr int Hh   = 32;
constexpr int KVHc = 8;
constexpr int HDc  = 128;
constexpr int MSc  = 4096;
constexpr int Sc   = SPc + Tt;     // 4096 (== MSc)
constexpr int REPc = Hh / KVHc;    // 4

typedef __attribute__((ext_vector_type(16))) _Float16 v16h;
typedef __attribute__((ext_vector_type(8)))  _Float16 v8h;
typedef __attribute__((ext_vector_type(8)))  float    v8f;

union H16 { v16h v; v8h h8[2]; _Float16 h[16]; };

__device__ __forceinline__ v8f wmma16(v16h a, v16h b, v8f c) {
  // D(16x16,f32) = A(16x32,f16) * B(32x16,f16) + C
  return __builtin_amdgcn_wmma_f32_16x16x32_f16(false, a, false, b, (short)0, c,
                                                false, false);
}

__device__ __forceinline__ void cvt16(H16& d, float4 f0, float4 f1, float4 f2, float4 f3) {
  d.h[0]=(_Float16)f0.x;  d.h[1]=(_Float16)f0.y;  d.h[2]=(_Float16)f0.z;  d.h[3]=(_Float16)f0.w;
  d.h[4]=(_Float16)f1.x;  d.h[5]=(_Float16)f1.y;  d.h[6]=(_Float16)f1.z;  d.h[7]=(_Float16)f1.w;
  d.h[8]=(_Float16)f2.x;  d.h[9]=(_Float16)f2.y;  d.h[10]=(_Float16)f2.z; d.h[11]=(_Float16)f2.w;
  d.h[12]=(_Float16)f3.x; d.h[13]=(_Float16)f3.y; d.h[14]=(_Float16)f3.z; d.h[15]=(_Float16)f3.w;
}

// =====================================================================
// Generic C = A(128 x 4096) * W(4096 x N) with WMMA f16.
// op==0: store f32 row-major to outf.
// op==1: fused RoPE + 1/sqrt(HD) scale, store f16 to q16 [b][h][t][hd].
// =====================================================================
__global__ __launch_bounds__(256)
void gqa_gemm_kernel(const float* __restrict__ A, const float* __restrict__ W,
                     int N, int op, float qscale,
                     const float* __restrict__ fcos, const float* __restrict__ fsin,
                     _Float16* __restrict__ q16, float* __restrict__ outf) {
  const int lane = threadIdx.x & 31;
  const int wave = threadIdx.x >> 5;
  const int n    = lane & 15;
  const int hs   = lane >> 4;
  const int col  = blockIdx.x * 128 + wave * 16 + n;

  v8f acc[8];
#pragma unroll
  for (int mt = 0; mt < 8; ++mt)
#pragma unroll
    for (int r = 0; r < 8; ++r) acc[mt][r] = 0.f;

  for (int kt = 0; kt < DIMc / 32; ++kt) {
    // B fragment: W[kt*32 + j + 16*hs][col], j = 0..15 (coalesces to 64B/line)
    H16 bf;
    const float* wp = W + (size_t)(kt * 32 + 16 * hs) * N + col;
    __builtin_prefetch(wp + (size_t)32 * N, 0, 0);   // global_prefetch_b8, next k-tile
#pragma unroll
    for (int j = 0; j < 16; ++j) bf.h[j] = (_Float16)wp[(size_t)j * N];

#pragma unroll
    for (int mt = 0; mt < 8; ++mt) {
      // A fragment: row = n, k(j) = j + 16*hs
      const float* ap = A + (size_t)(mt * 16 + n) * DIMc + kt * 32 + 16 * hs;
      float4 f0 = *(const float4*)(ap);
      float4 f1 = *(const float4*)(ap + 4);
      float4 f2 = *(const float4*)(ap + 8);
      float4 f3 = *(const float4*)(ap + 12);
      H16 af; cvt16(af, f0, f1, f2, f3);
      acc[mt] = wmma16(af.v, bf.v, acc[mt]);
    }
  }

  if (op == 0) {
#pragma unroll
    for (int mt = 0; mt < 8; ++mt)
#pragma unroll
      for (int r = 0; r < 8; ++r)
        outf[(size_t)(mt * 16 + r + 8 * hs) * N + col] = acc[mt][r];
  } else {
    // q path: RoPE (pair mix across adjacent lanes) + scale + f16 store
    const int head = col >> 7;
    const int hd   = col & 127;
    const int fi   = hd >> 1;
    const bool ev  = (hd & 1) == 0;
#pragma unroll
    for (int mt = 0; mt < 8; ++mt) {
#pragma unroll
      for (int r = 0; r < 8; ++r) {
        const int t = r + 8 * hs;
        float v = acc[mt][r];
        float p = __shfl_xor(v, 1);          // partner hd (hd parity == lane parity)
        float c = fcos[t * (HDc / 2) + fi];
        float s = fsin[t * (HDc / 2) + fi];
        float res = ev ? (v * c - p * s) : (p * s + v * c);
        res *= qscale;
        q16[(((size_t)(mt * Hh + head)) * Tt + t) * HDc + hd] = (_Float16)res;
      }
    }
  }
}

// =====================================================================
// Flash attention, one block per (b, h). 8 waves x 512 keys each with
// online softmax; V-chunks staged to LDS via async global->LDS copies
// (ASYNCcnt) and reused by both the score pass and the P*V pass.
// NOTE: per the reference, BOTH keys and values come from cache_v/vnew.
// =====================================================================
union AttnSmem {
  float vch[8][32][128];    // 128 KB: per-wave staged V chunk (main loop)
  float comb[8][16][128];   //  64 KB: per-wave partial outputs (after loop)
};

__global__ __launch_bounds__(256)
void gqa_attn_kernel(const _Float16* __restrict__ q16,
                     const float* __restrict__ cache_v,
                     const float* __restrict__ vnew,
                     const float* __restrict__ mask,
                     float* __restrict__ attn_out) {
  __shared__ AttnSmem smem;
  __shared__ _Float16 plds[8][16][32];   // per-wave P-tile bounce (8 KB)
  __shared__ float    mlm[8][16];
  __shared__ float    mll[8][16];

  const int b    = blockIdx.x / Hh;
  const int h    = blockIdx.x % Hh;
  const int kvh  = h / REPc;
  const int lane = threadIdx.x & 31;
  const int wave = threadIdx.x >> 5;
  const int n    = lane & 15;
  const int hs   = lane >> 4;

  // Q A-fragments: 4 k-tiles over HD=128, two 16B loads each
  v16h qa[4];
  {
    const _Float16* qp = q16 + (((size_t)(b * Hh + h)) * Tt + n) * HDc;
#pragma unroll
    for (int kt = 0; kt < 4; ++kt) {
      H16 af;
      af.h8[0] = *(const v8h*)(qp + kt * 32 + 16 * hs);
      af.h8[1] = *(const v8h*)(qp + kt * 32 + 16 * hs + 8);
      qa[kt] = af.v;
    }
  }

  float m_run[8], l_run[8];
  v8f acc[8];
#pragma unroll
  for (int r = 0; r < 8; ++r) { m_run[r] = -3.0e38f; l_run[r] = 0.f; }
#pragma unroll
  for (int nt = 0; nt < 8; ++nt)
#pragma unroll
    for (int r = 0; r < 8; ++r) acc[nt][r] = 0.f;

  const int sbase = wave * (Sc / 8);   // 512 keys per wave

  for (int ch = 0; ch < 16; ++ch) {    // 16 chunks x 32 keys
    const int s0 = sbase + ch * 32;

    // ---- async-stage the 32x128 f32 V-chunk into this wave's LDS region ----
    {
#pragma unroll
      for (int i = 0; i < 32; ++i) {
        const int slot = i * 32 + lane;   // 1024 float4 slots per chunk
        const int krow = slot >> 5;       // local key 0..31
        const int c4   = slot & 31;       // float4 within the 128-float row
        const int key  = s0 + krow;
        const float* gp = (key < SPc)
          ? cache_v + (((size_t)(b * MSc + key)) * KVHc + kvh) * HDc + c4 * 4
          : vnew    + (((size_t)(b * Tt + (key - SPc))) * KVHc + kvh) * HDc + c4 * 4;
        const unsigned lds_off = (unsigned)(uintptr_t)(&smem.vch[wave][krow][c4 * 4]);
        const unsigned long long ga = (unsigned long long)(uintptr_t)gp;
        asm volatile("global_load_async_to_lds_b128 %0, %1, off"
                     :: "v"(lds_off), "v"(ga) : "memory");
      }
      asm volatile("s_wait_asynccnt 0x0" ::: "memory");
    }

    // ---- scores for two 16-key tiles (B fragments from LDS) ----
    v8f sc[2];
#pragma unroll
    for (int sub = 0; sub < 2; ++sub) {
      const int key = s0 + sub * 16 + n;
      v8f c;
#pragma unroll
      for (int r = 0; r < 8; ++r) c[r] = 0.f;
      const float* kp = &smem.vch[wave][sub * 16 + n][0];
#pragma unroll
      for (int kt = 0; kt < 4; ++kt) {
        const float* p4 = kp + kt * 32 + 16 * hs;   // K = hd, contiguous per lane
        H16 bf;
        cvt16(bf, *(const float4*)(p4), *(const float4*)(p4 + 4),
                  *(const float4*)(p4 + 8), *(const float4*)(p4 + 12));
        c = wmma16(qa[kt], bf.v, c);
      }
#pragma unroll
      for (int r = 0; r < 8; ++r) c[r] += mask[(size_t)(r + 8 * hs) * Sc + key];
      sc[sub] = c;
    }

    // ---- online softmax update (rows live in 16-lane halves) ----
    float p0[8], p1[8];
#pragma unroll
    for (int r = 0; r < 8; ++r) {
      float tm = fmaxf(sc[0][r], sc[1][r]);
#pragma unroll
      for (int mb = 1; mb <= 8; mb <<= 1) tm = fmaxf(tm, __shfl_xor(tm, mb));
      float mnew = fmaxf(m_run[r], tm);
      float corr = __expf(m_run[r] - mnew);
      l_run[r] *= corr;
#pragma unroll
      for (int nt = 0; nt < 8; ++nt) acc[nt][r] *= corr;
      p0[r] = __expf(sc[0][r] - mnew);
      p1[r] = __expf(sc[1][r] - mnew);
      float ps = p0[r] + p1[r];
#pragma unroll
      for (int mb = 1; mb <= 8; mb <<= 1) ps += __shfl_xor(ps, mb);
      l_run[r] += ps;
      m_run[r] = mnew;
    }

    // ---- P tile: C-layout -> A-layout via per-wave LDS (in-order LDS per wave) ----
#pragma unroll
    for (int r = 0; r < 8; ++r) {
      plds[wave][r + 8 * hs][n]      = (_Float16)p0[r];
      plds[wave][r + 8 * hs][16 + n] = (_Float16)p1[r];
    }
    __threadfence_block();
    H16 pf;
    pf.h8[0] = *(const v8h*)&plds[wave][n][16 * hs];
    pf.h8[1] = *(const v8h*)&plds[wave][n][16 * hs + 8];

    // ---- P x V: 8 N-tiles over HD=128, B fragments from LDS ----
#pragma unroll
    for (int nt = 0; nt < 8; ++nt) {
      const int hd = nt * 16 + n;
      H16 bf;
#pragma unroll
      for (int j = 0; j < 16; ++j)
        bf.h[j] = (_Float16)smem.vch[wave][16 * hs + j][hd];
      acc[nt] = wmma16(pf.v, bf.v, acc[nt]);
    }
  }

  // vch and comb overlap across waves: everyone must leave the main loop first
  __syncthreads();

  // ---- combine the 8 wave partials ----
#pragma unroll
  for (int nt = 0; nt < 8; ++nt)
#pragma unroll
    for (int r = 0; r < 8; ++r)
      smem.comb[wave][r + 8 * hs][nt * 16 + n] = acc[nt][r];
  if (n == 0) {
#pragma unroll
    for (int r = 0; r < 8; ++r) {
      mlm[wave][r + 8 * hs] = m_run[r];
      mll[wave][r + 8 * hs] = l_run[r];
    }
  }
  __syncthreads();

  for (int idx = threadIdx.x; idx < Tt * HDc; idx += 256) {
    const int t  = idx >> 7;
    const int hd = idx & 127;
    float M = -3.0e38f;
#pragma unroll
    for (int w = 0; w < 8; ++w) M = fmaxf(M, mlm[w][t]);
    float L = 0.f, O = 0.f;
#pragma unroll
    for (int w = 0; w < 8; ++w) {
      float e = __expf(mlm[w][t] - M);
      L += e * mll[w][t];
      O += e * smem.comb[w][t][hd];
    }
    attn_out[(((size_t)(b * Tt + t)) * Hh + h) * HDc + hd] = O / L;
  }
}

// =====================================================================
extern "C" void kernel_launch(void* const* d_in, const int* in_sizes, int n_in,
                              void* d_out, int out_size, void* d_ws, size_t ws_size,
                              hipStream_t stream) {
  (void)in_sizes; (void)n_in; (void)out_size; (void)ws_size;
  const float* x       = (const float*)d_in[0];
  const float* wq      = (const float*)d_in[1];
  // d_in[2] = wk, d_in[5] = cache_k: provably dead in the reference (kk uses v_full)
  const float* wv      = (const float*)d_in[3];
  const float* wproj   = (const float*)d_in[4];
  const float* cache_v = (const float*)d_in[6];
  const float* fcos    = (const float*)d_in[7];
  const float* fsin    = (const float*)d_in[8];
  const float* maskp   = (const float*)d_in[9];
  // d_in[10] = start_pos (== SPc, fixed by the reference)

  char* ws = (char*)d_ws;
  _Float16* q16  = (_Float16*)ws;                                   // 1 MB
  float*    vnew = (float*)(ws + (1u << 20));                       // 0.5 MB
  float*    attn = (float*)(ws + (1u << 20) + (1u << 19));          // 2 MB

  const float qscale = 0.08838834764831845f;   // 1/sqrt(128)

  // q = rope(x @ wq) * scale   (f16, [b][h][t][hd])
  gqa_gemm_kernel<<<dim3(Hh * HDc / 128), dim3(256), 0, stream>>>(
      x, wq, Hh * HDc, 1, qscale, fcos, fsin, q16, nullptr);
  // vnew = x @ wv              (f32, [b][t][kvh][hd] == row-major [m][1024])
  gqa_gemm_kernel<<<dim3(KVHc * HDc / 128), dim3(256), 0, stream>>>(
      x, wv, KVHc * HDc, 0, 1.f, nullptr, nullptr, nullptr, vnew);
  // attention (K and V both from cache_v/vnew, per reference)
  gqa_attn_kernel<<<dim3(Bb * Hh), dim3(256), 0, stream>>>(
      q16, cache_v, vnew, maskp, attn);
  // out = attn @ wproj
  gqa_gemm_kernel<<<dim3(DIMc / 128), dim3(256), 0, stream>>>(
      attn, wproj, DIMc, 0, 1.f, nullptr, nullptr, nullptr, (float*)d_out);
}
